// SSIM3D_36893769072654
// MI455X (gfx1250) — matrix operand order repaired and checked by reading, not compiled, not run on hardware
//
#include <hip/hip_runtime.h>

// ---------------------------------------------------------------------------
// SSIM3D for MI455X (gfx1250, wave32).
// Separable 11-tap Gaussian blur done as banded-matrix x data-tile products on
// the matrix cores: v_wmma_f32_16x16x32_f16 (fp16 data/weights, fp32 accum).
// 3 passes (W, H, D+SSIM+reduce) with fp16 intermediates to minimize HBM.
// ---------------------------------------------------------------------------

typedef __attribute__((ext_vector_type(16))) _Float16 v16h;
typedef __attribute__((ext_vector_type(8)))  _Float16 v8h;
typedef __attribute__((ext_vector_type(8)))  float    v8f;

#define C_   2
#define D_   150
#define H_   214
#define W_   214
#define HP   224          // padded H for intermediate buffers
#define WP   224          // padded W for intermediate buffers
#define CROP 21
#define RD   192
#define RH   256
#define RW   256
#define NWCH 14           // ceil(214/16) chunks along W
#define NHT  14           // ceil(214/16) tiles along H
#define NDT  10           // ceil(150/16) tiles along D

#define HS   WP                       // h stride (halfs)
#define DSTR (HP*WP)                  // d stride
#define FS_  ((size_t)C_*D_*HP*WP)    // per-field stride = 15,052,800 halfs

__device__ __forceinline__ void lds_sync() {
  // Same-wave LDS ordering is in-order per the ISA; this is a hard fence for
  // the compiler plus an explicit DS-counter drain.
  asm volatile("s_wait_dscnt 0" ::: "memory");
}

// Banded Gaussian A-matrix (16x32 f16), contiguous K per lane half:
// lanes 0-15 hold K = 0..15, lanes 16-31 hold K = 16..31.
__device__ __forceinline__ v16h make_bandA(int lane) {
  const float G[11] = {0.00102849f, 0.00759855f, 0.03600069f, 0.10936069f,
                       0.21300528f, 0.26601159f, 0.21300528f, 0.10936069f,
                       0.03600069f, 0.00759855f, 0.00102849f};
  int M = lane & 15;
  v16h a;
#pragma unroll
  for (int i = 0; i < 16; ++i) {
    int K = (lane & 16) + i;
    int t = K - M;
    a[i] = (t >= 0 && t <= 10) ? (_Float16)G[t] : (_Float16)0.0f;
  }
  return a;
}

// ---------------------------------------------------------------------------
// Pass 1: blur along W.  B[k][n]: k = input w position, n = h line.
// Stage 16h x 32w of all five fields (fp16) in per-wave LDS, then 5 WMMAs.
// Output D[m][n]: m = output w, n = h -> one packed 16B store per lane/field.
// ---------------------------------------------------------------------------
__global__ __launch_bounds__(256) void k_blurW(const float* __restrict__ raw,
                                               const float* __restrict__ dst,
                                               _Float16* __restrict__ out) {
  __shared__ alignas(64) _Float16 tile[8][5][16][32];
  int tid = threadIdx.x, wv = tid >> 5, lane = tid & 31;
  int bid = blockIdx.x;
  int ht = bid % NHT;
  int d  = (bid / NHT) % D_;
  int cc = bid / (NHT * D_);
  int h0 = ht * 16;
  v16h A = make_bandA(lane);

  const size_t inBase = ((size_t)cc * RD + (size_t)(d + CROP)) * RH;

  for (int ch = wv; ch < NWCH; ch += 8) {
    int w0 = ch * 16 - 5;                          // K=0 input position
    for (int r = 0; r < 16; ++r) {                 // stage one h-row per iter
      int hh = h0 + r;                             // rows >= H_ are junk, unused
      int wi = w0 + lane;
      float rv = 0.f, dv = 0.f;
      if (wi >= 0 && wi < W_) {                    // SAME zero padding along W
        size_t idx = (inBase + (size_t)(hh + CROP)) * RW + (size_t)(wi + CROP);
        rv = raw[idx];
        dv = dst[idx];
      }
      tile[wv][0][r][lane] = (_Float16)rv;
      tile[wv][1][r][lane] = (_Float16)dv;
      tile[wv][2][r][lane] = (_Float16)(rv * rv);
      tile[wv][3][r][lane] = (_Float16)(dv * dv);
      tile[wv][4][r][lane] = (_Float16)(rv * dv);
    }
    lds_sync();

    int n    = lane & 15;
    int kb   = lane & 16;                          // K half selected by lane group
    int hout = h0 + n;
    size_t obase = ((size_t)cc * D_ + d) * (size_t)DSTR +
                   (size_t)hout * HS + (size_t)(ch * 16) + ((lane & 16) ? 8 : 0);
#pragma unroll
    for (int f = 0; f < 5; ++f) {
      v16h b = *(const v16h*)&tile[wv][f][n][kb];
      v8f acc = {};
      acc = __builtin_amdgcn_wmma_f32_16x16x32_f16(false, A, false, b,
                                                   (short)0, acc, false, false);
      v8h o;
#pragma unroll
      for (int j = 0; j < 8; ++j) o[j] = (_Float16)acc[j];
      *(v8h*)(out + (size_t)f * FS_ + obase) = o;  // global_store_b128
    }
    lds_sync();
  }
}

// ---------------------------------------------------------------------------
// Pass 2: blur along H.  B[k][n]: k = input h position, n = w.
// Lane L loads row h0+L (16 contiguous halfs in w), transposes through LDS.
// ---------------------------------------------------------------------------
__global__ __launch_bounds__(256) void k_blurH(const _Float16* __restrict__ in,
                                               _Float16* __restrict__ out) {
  __shared__ alignas(64) _Float16 tile[8][5][16][32];
  int tid = threadIdx.x, wv = tid >> 5, lane = tid & 31;
  int bid = blockIdx.x;
  int ht = bid % NHT;
  int d  = (bid / NHT) % D_;
  int cc = bid / (NHT * D_);
  int h0 = ht * 16 - 5;
  v16h A = make_bandA(lane);
  size_t cdbase = ((size_t)cc * D_ + d) * (size_t)DSTR;

  for (int ch = wv; ch < NWCH; ch += 8) {
    int wb  = ch * 16;
    int hin = h0 + lane;                           // this lane's K row
    bool ok = (hin >= 0 && hin < H_);              // SAME zero padding along H
#pragma unroll
    for (int f = 0; f < 5; ++f) {
      v16h row = {};
      if (ok)
        row = *(const v16h*)(in + (size_t)f * FS_ + cdbase +
                             (size_t)hin * HS + (size_t)wb);
#pragma unroll
      for (int n2 = 0; n2 < 16; ++n2) tile[wv][f][n2][lane] = row[n2];
    }
    lds_sync();

    int n = lane & 15, kb = lane & 16;
    int w = wb + n;
    int mof = (lane & 16) ? 8 : 0;
#pragma unroll
    for (int f = 0; f < 5; ++f) {
      v16h b = *(const v16h*)&tile[wv][f][n][kb];
      v8f acc = {};
      acc = __builtin_amdgcn_wmma_f32_16x16x32_f16(false, A, false, b,
                                                   (short)0, acc, false, false);
#pragma unroll
      for (int j = 0; j < 8; ++j) {
        int hout = ht * 16 + mof + j;
        if (hout < H_)
          out[(size_t)f * FS_ + cdbase + (size_t)hout * HS + (size_t)w] =
              (_Float16)acc[j];
      }
    }
    lds_sync();
  }
}

// ---------------------------------------------------------------------------
// Pass 3: blur along D fused with SSIM map + deterministic block reduction.
// ---------------------------------------------------------------------------
__global__ __launch_bounds__(256) void k_blurD_ssim(const _Float16* __restrict__ in,
                                                    float* __restrict__ partial) {
  __shared__ alignas(64) _Float16 tile[8][5][16][32];
  __shared__ float red[256];
  int tid = threadIdx.x, wv = tid >> 5, lane = tid & 31;
  int bid = blockIdx.x;
  int h  = bid % H_;
  int dt = (bid / H_) % NDT;
  int cc = bid / (H_ * NDT);
  int d0 = dt * 16 - 5;
  v16h A = make_bandA(lane);
  const float c1 = 6.5025f, c2 = 58.5225f;

  float sacc = 0.f;
  for (int ch = wv; ch < NWCH; ch += 8) {
    int wb  = ch * 16;
    int din = d0 + lane;
    bool ok = (din >= 0 && din < D_);              // SAME zero padding along D
#pragma unroll
    for (int f = 0; f < 5; ++f) {
      v16h row = {};
      if (ok)
        row = *(const v16h*)(in + (size_t)f * FS_ +
                             ((size_t)cc * D_ + (size_t)din) * (size_t)DSTR +
                             (size_t)h * HS + (size_t)wb);
#pragma unroll
      for (int n2 = 0; n2 < 16; ++n2) tile[wv][f][n2][lane] = row[n2];
    }
    lds_sync();

    int n = lane & 15, kb = lane & 16;
    int w = wb + n;
    int mof = (lane & 16) ? 8 : 0;
    v8f r[5];
#pragma unroll
    for (int f = 0; f < 5; ++f) {
      v16h b = *(const v16h*)&tile[wv][f][n][kb];
      v8f acc = {};
      r[f] = __builtin_amdgcn_wmma_f32_16x16x32_f16(false, A, false, b,
                                                    (short)0, acc, false, false);
    }
    if (w < W_) {
#pragma unroll
      for (int j = 0; j < 8; ++j) {
        int dout = dt * 16 + mof + j;
        if (dout < D_) {
          float mu1 = r[0][j], mu2 = r[1][j];
          float mu1s = mu1 * mu1, mu2s = mu2 * mu2, mu12 = mu1 * mu2;
          float v1 = r[2][j] - mu1s, v2 = r[3][j] - mu2s, cv = r[4][j] - mu12;
          sacc += ((2.f * mu12 + c1) * (2.f * cv + c2)) /
                  ((mu1s + mu2s + c1) * (v1 + v2 + c2));
        }
      }
    }
    lds_sync();
  }

  red[tid] = sacc;
  __syncthreads();
  for (int s = 128; s > 0; s >>= 1) {
    if (tid < s) red[tid] += red[tid + s];
    __syncthreads();
  }
  if (tid == 0) partial[bid] = red[0];
}

// Deterministic final reduce: fixed partition + fixed-order tree.
__global__ __launch_bounds__(512) void k_final(const float* __restrict__ partial,
                                               float* __restrict__ outv) {
  __shared__ float red[512];
  int tid = threadIdx.x;
  int cc = tid >> 8;            // channel 0: threads 0-255, channel 1: 256-511
  int t  = tid & 255;
  float acc = 0.f;
  for (int i = t; i < NDT * H_; i += 256) acc += partial[cc * (NDT * H_) + i];
  red[tid] = acc;
  __syncthreads();
  for (int s = 128; s > 0; s >>= 1) {
    if (t < s) red[tid] += red[tid + s];
    __syncthreads();
  }
  if (t == 0) outv[cc] = red[tid] / (float)((size_t)D_ * H_ * W_);
}

extern "C" void kernel_launch(void* const* d_in, const int* in_sizes, int n_in,
                              void* d_out, int out_size, void* d_ws, size_t ws_size,
                              hipStream_t stream) {
  (void)in_sizes; (void)n_in; (void)out_size; (void)ws_size;
  const float* raw = (const float*)d_in[0];
  const float* dst = (const float*)d_in[1];

  _Float16* buf1 = (_Float16*)d_ws;             // W-blurred fields (fp16)
  _Float16* buf2 = buf1 + 5 * FS_;              // W+H-blurred fields (fp16)
  float* partial = (float*)(buf2 + 5 * FS_);    // per-block SSIM partial sums

  k_blurW<<<C_ * D_ * NHT, 256, 0, stream>>>(raw, dst, buf1);
  k_blurH<<<C_ * D_ * NHT, 256, 0, stream>>>(buf1, buf2);
  k_blurD_ssim<<<C_ * NDT * H_, 256, 0, stream>>>(buf2, partial);
  k_final<<<1, 512, 0, stream>>>(partial, (float*)d_out);
}
